// NeuronGPTOSSBlock_86320252715717
// MI455X (gfx1250) — compile-verified, matrix operand order checked by reading, not run on hardware
//
#include <hip/hip_runtime.h>
#include <hip/hip_bf16.h>

// ---------------- constants (match reference) ----------------
#define Hd   2880     // hidden size
#define NQ   64
#define NKV  8
#define Dh   64
#define NQD  (NQ*Dh)    // 4096
#define KVD  (NKV*Dh)   // 512
#define NE   8
#define Iin  2880

typedef __attribute__((ext_vector_type(16))) _Float16 v16h;
typedef __attribute__((ext_vector_type(8)))  float    v8f;
typedef __attribute__((ext_vector_type(4)))  int      v4i;

union HV { v16h v; _Float16 h[16]; };

// K-index (0..31) inside a 16x16x32 f16 WMMA fragment for element e, lane-group lg.
// Per CDNA5 ISA 7.12.2: lanes 0-15 hold K {0..7, 16..23}, lanes 16-31 hold K {8..15, 24..31}.
__device__ __forceinline__ int frag_k(int e, int lg) {
    return (e < 8) ? (8 * lg + e) : (16 + 8 * lg + (e - 8));
}

// ---- gfx1250 async global->LDS copy (ASYNCcnt), with safe fallback ----
#if defined(__HIP_DEVICE_COMPILE__) && \
    __has_builtin(__builtin_amdgcn_global_load_async_to_lds_b128) && \
    __has_builtin(__builtin_amdgcn_s_wait_asynccnt)
#define USE_ASYNC 1
#else
#define USE_ASYNC 0
#endif

#define AS1 __attribute__((address_space(1)))
#define AS3 __attribute__((address_space(3)))

#if USE_ASYNC
#define ASYNC_CP16(gsrc, ldst) \
    __builtin_amdgcn_global_load_async_to_lds_b128((AS1 v4i*)(gsrc), (AS3 v4i*)(ldst), 0, 0)
#define WAIT_ASYNC(n) __builtin_amdgcn_s_wait_asynccnt(n)
#else
#define ASYNC_CP16(gsrc, ldst) (*(float4*)(ldst) = *(const float4*)(gsrc))
#define WAIT_ASYNC(n) do {} while (0)
#endif

// ---------------- generic WMMA GEMM: C[M,N] = A[M,K] * B[K,N] + bias ----------------
// Block tile 128x64, 8 waves (256 threads), each wave computes a 32x32 quadrant.
// Double-buffered f32 tiles in LDS filled by async global->LDS b128 copies; f32->f16
// conversion happens at fragment-gather time (async DMA moves raw bytes).
#define BM 128
#define BN 64
#define BK 32
#define AKP (BK + 8)   // 40 floats: 160B row stride (16B aligned chunks)
#define BNP (BN + 8)   // 72 floats: 288B row stride

__global__ __launch_bounds__(256)
void gemm_wmma_kernel(const float* __restrict__ A, const float* __restrict__ B,
                      const float* __restrict__ bias, float* __restrict__ C,
                      int M, int N, int K)
{
    __shared__ __align__(16) float Asf[2][BM][AKP];
    __shared__ __align__(16) float Bsf[2][BK][BNP];

    const int tid  = threadIdx.x;
    const int lane = tid & 31;
    const int wave = tid >> 5;            // 0..7
    const int lg   = lane >> 4;
    const int li   = lane & 15;

    const int m0 = blockIdx.y * BM;
    const int n0 = blockIdx.x * BN;
    const int wrow = (wave >> 1) * 32;    // 0,32,64,96
    const int wcol = (wave & 1)  * 32;    // 0,32

    // Issue one tile-pair of async copies into buffer `buf` for k-offset k0.
    // A tile: 128x32 f32 = 1024 x 16B chunks -> 4 rounds of 256 threads.
    // B tile:  32x64 f32 =  512 x 16B chunks -> 2 rounds. Total 6 ops/thread.
    auto stage = [&](int buf, int k0) {
        #pragma unroll
        for (int i = 0; i < (BM * BK / 4) / 256; ++i) {
            const int c   = tid + i * 256;
            const int row = c >> 3;            // 8 chunks per 32-float row
            const int off = (c & 7) * 4;
            ASYNC_CP16(A + (size_t)(m0 + row) * K + k0 + off, &Asf[buf][row][off]);
        }
        #pragma unroll
        for (int i = 0; i < (BK * BN / 4) / 256; ++i) {
            const int c   = tid + i * 256;
            const int row = c >> 4;            // 16 chunks per 64-float row
            const int off = (c & 15) * 4;
            ASYNC_CP16(B + (size_t)(k0 + row) * N + n0 + off, &Bsf[buf][row][off]);
        }
    };

    v8f acc[2][2] = {};
    const int nK = K / BK;

    stage(0, 0);
    for (int kt = 0; kt < nK; ++kt) {
        const int cb = kt & 1;
        if (kt + 1 < nK) {
            stage((kt + 1) & 1, (kt + 1) * BK);
            WAIT_ASYNC(6);        // 6 newest (next tile) may remain in flight
        } else {
            WAIT_ASYNC(0);
        }
        __syncthreads();          // all waves' copies for buffer cb complete

        HV a[2], b[2];
        #pragma unroll
        for (int mt = 0; mt < 2; ++mt)
            #pragma unroll
            for (int e = 0; e < 16; ++e)
                a[mt].h[e] = (_Float16)Asf[cb][wrow + mt * 16 + li][frag_k(e, lg)];
        #pragma unroll
        for (int nt = 0; nt < 2; ++nt)
            #pragma unroll
            for (int e = 0; e < 16; ++e)
                b[nt].h[e] = (_Float16)Bsf[cb][frag_k(e, lg)][wcol + nt * 16 + li];

        #pragma unroll
        for (int mt = 0; mt < 2; ++mt)
            #pragma unroll
            for (int nt = 0; nt < 2; ++nt)
                acc[mt][nt] = __builtin_amdgcn_wmma_f32_16x16x32_f16(
                    false, a[mt].v, false, b[nt].v, (short)0, acc[mt][nt], false, false);

        __syncthreads();          // protect buffer cb before it is re-staged
    }

    #pragma unroll
    for (int mt = 0; mt < 2; ++mt)
        #pragma unroll
        for (int nt = 0; nt < 2; ++nt) {
            const int col = n0 + wcol + nt * 16 + li;
            const float bv = bias ? bias[col] : 0.0f;
            #pragma unroll
            for (int r = 0; r < 8; ++r) {
                const int row = m0 + wrow + mt * 16 + 8 * lg + r;
                C[(size_t)row * N + col] = acc[mt][nt][r] + bv;
            }
        }
}

// ---------------- flash-style attention: one wave per (16-query block, head) ----------------
__global__ __launch_bounds__(32)
void attn_kernel(const float* __restrict__ q, const float* __restrict__ k,
                 const float* __restrict__ v, float* __restrict__ ctx, int T)
{
    __shared__ _Float16 Ps[16][32 + 8];

    const int lane = threadIdx.x & 31;
    const int lg = lane >> 4, li = lane & 15;
    const int qb = blockIdx.x;      // query block of 16 rows
    const int hq = blockIdx.y;      // query head
    const int hk = hq >> 3;         // GQA: NQ/NKV = 8
    const int r0 = qb * 16;
    const float scale = 0.125f;     // 1/sqrt(64)

    // Q fragments (pre-scaled): A-layout, two K=32 chunks over head dim 64
    HV qa[2];
    {
        const float* qr = q + (size_t)(r0 + li) * NQD + hq * Dh;
        #pragma unroll
        for (int kc = 0; kc < 2; ++kc)
            #pragma unroll
            for (int e = 0; e < 16; ++e)
                qa[kc].h[e] = (_Float16)(qr[kc * 32 + frag_k(e, lg)] * scale);
    }

    v8f o[4] = {};
    float mrow[8], lrow[8];
    #pragma unroll
    for (int r = 0; r < 8; ++r) { mrow[r] = -1e30f; lrow[r] = 0.0f; }

    const int nChunks = (r0 + 16 + 31) / 32;     // 32-key chunks up to the diagonal
    for (int ch = 0; ch < nChunks; ++ch) {
        const int c0 = ch * 32;

        // scores S = Q * K^T for this 16x32 chunk (2 n-tiles x 2 K-steps)
        v8f sc[2] = {};
        #pragma unroll
        for (int nt = 0; nt < 2; ++nt) {
            const int key = c0 + nt * 16 + li;
            const bool ok = key < T;
            const float* kr = k + (size_t)(ok ? key : 0) * KVD + hk * Dh;
            #pragma unroll
            for (int kc = 0; kc < 2; ++kc) {
                HV kb;
                #pragma unroll
                for (int e = 0; e < 16; ++e)
                    kb.h[e] = (_Float16)(ok ? kr[kc * 32 + frag_k(e, lg)] : 0.0f);
                sc[nt] = __builtin_amdgcn_wmma_f32_16x16x32_f16(
                    false, qa[kc].v, false, kb.v, (short)0, sc[nt], false, false);
            }
        }

        // causal mask + online softmax (row = 8*lg + r, cols striped over 16 lanes)
        #pragma unroll
        for (int r = 0; r < 8; ++r) {
            const int qrow = r0 + 8 * lg + r;
            const int key0 = c0 + li, key1 = c0 + 16 + li;
            float s0 = (key0 <= qrow) ? sc[0][r] : -1e30f;
            float s1 = (key1 <= qrow) ? sc[1][r] : -1e30f;
            float t = fmaxf(s0, s1);
            #pragma unroll
            for (int off = 1; off < 16; off <<= 1) t = fmaxf(t, __shfl_xor(t, off, 32));
            const float mnew = fmaxf(mrow[r], t);
            const float sclr = __expf(mrow[r] - mnew);
            const float e0 = __expf(s0 - mnew);
            const float e1 = __expf(s1 - mnew);
            float rs = e0 + e1;
            #pragma unroll
            for (int off = 1; off < 16; off <<= 1) rs += __shfl_xor(rs, off, 32);
            lrow[r] = lrow[r] * sclr + rs;
            mrow[r] = mnew;
            #pragma unroll
            for (int j = 0; j < 4; ++j) o[j][r] *= sclr;
            Ps[8 * lg + r][li]      = (_Float16)e0;
            Ps[8 * lg + r][16 + li] = (_Float16)e1;
        }
        __syncthreads();

        // O += P(16x32) * V(32x64)
        HV pa;
        #pragma unroll
        for (int e = 0; e < 16; ++e) pa.h[e] = Ps[li][frag_k(e, lg)];
        #pragma unroll
        for (int j = 0; j < 4; ++j) {
            HV vb;
            #pragma unroll
            for (int e = 0; e < 16; ++e) {
                const int key = c0 + frag_k(e, lg);
                vb.h[e] = (_Float16)((key < T) ? v[(size_t)key * KVD + hk * Dh + j * 16 + li] : 0.0f);
            }
            o[j] = __builtin_amdgcn_wmma_f32_16x16x32_f16(
                false, pa.v, false, vb.v, (short)0, o[j], false, false);
        }
        __syncthreads();
    }

    #pragma unroll
    for (int j = 0; j < 4; ++j)
        #pragma unroll
        for (int r = 0; r < 8; ++r) {
            const int row = r0 + 8 * lg + r;
            ctx[(size_t)row * NQD + hq * Dh + j * 16 + li] = o[j][r] / lrow[r];
        }
}

// ---------------- small VALU kernels ----------------
__global__ __launch_bounds__(256)
void rmsnorm_kernel(const float* __restrict__ x, const float* __restrict__ w,
                    float* __restrict__ out)
{
    __shared__ float red[8];
    const int t = blockIdx.x;
    const float* xr = x + (size_t)t * Hd;
    float s = 0.0f;
    for (int h = threadIdx.x; h < Hd; h += 256) { float val = xr[h]; s += val * val; }
    #pragma unroll
    for (int off = 1; off < 32; off <<= 1) s += __shfl_xor(s, off, 32);
    if ((threadIdx.x & 31) == 0) red[threadIdx.x >> 5] = s;
    __syncthreads();
    if (threadIdx.x < 8) {
        float vsum = red[threadIdx.x];
        #pragma unroll
        for (int off = 1; off < 8; off <<= 1) vsum += __shfl_xor(vsum, off, 32);
        if (threadIdx.x == 0) red[0] = vsum;
    }
    __syncthreads();
    const float inv = rsqrtf(red[0] / (float)Hd + 1e-5f);
    for (int h = threadIdx.x; h < Hd; h += 256)
        out[(size_t)t * Hd + h] = xr[h] * inv * w[h];
}

__global__ void rope_kernel(float* __restrict__ q, float* __restrict__ k,
                            const int* __restrict__ pos)
{
    const int t = blockIdx.x;
    const float p = (float)pos[t];
    for (int i = threadIdx.x; i < (NQ + NKV) * 32; i += blockDim.x) {
        const int head = i >> 5;
        const int idx  = i & 31;
        const float inv_freq = __powf(150000.0f, -(float)(2 * idx) / 64.0f);
        float s, c;
        __sincosf(p * inv_freq, &s, &c);
        float* base = (head < NQ) ? (q + (size_t)t * NQD + head * Dh)
                                  : (k + (size_t)t * KVD + (head - NQ) * Dh);
        const float x1 = base[idx], x2 = base[idx + 32];
        base[idx]      = x1 * c - x2 * s;
        base[idx + 32] = x2 * c + x1 * s;
    }
}

__global__ __launch_bounds__(256)
void router_kernel(const float* __restrict__ tn, const float* __restrict__ rwt,
                   float* __restrict__ rw)
{
    __shared__ float sh[8][NE];
    const int t = blockIdx.x;
    float acc[NE] = {};
    const float* xr = tn + (size_t)t * Hd;
    for (int h = threadIdx.x; h < Hd; h += 256) {
        const float x = xr[h];
        #pragma unroll
        for (int e = 0; e < NE; ++e) acc[e] += x * rwt[(size_t)h * NE + e];
    }
    #pragma unroll
    for (int e = 0; e < NE; ++e)
        #pragma unroll
        for (int off = 1; off < 32; off <<= 1) acc[e] += __shfl_xor(acc[e], off, 32);
    if ((threadIdx.x & 31) == 0)
        for (int e = 0; e < NE; ++e) sh[threadIdx.x >> 5][e] = acc[e];
    __syncthreads();
    if (threadIdx.x == 0) {
        float logit[NE];
        for (int e = 0; e < NE; ++e) {
            float s = 0.0f;
            for (int wv = 0; wv < 8; ++wv) s += sh[wv][e];
            logit[e] = s;
        }
        float mx = logit[0];
        for (int e = 1; e < NE; ++e) mx = fmaxf(mx, logit[e]);
        float den = 0.0f, aff[NE];
        for (int e = 0; e < NE; ++e) { aff[e] = __expf(logit[e] - mx); den += aff[e]; }
        for (int e = 0; e < NE; ++e) aff[e] /= den;
        int i1 = 0;
        for (int e = 1; e < NE; ++e) if (aff[e] > aff[i1]) i1 = e;
        int i2 = (i1 == 0) ? 1 : 0;
        for (int e = 0; e < NE; ++e) if (e != i1 && aff[e] > aff[i2]) i2 = e;
        const float s2 = aff[i1] + aff[i2];
        for (int e = 0; e < NE; ++e) rw[(size_t)t * NE + e] = 0.0f;
        rw[(size_t)t * NE + i1] = aff[i1] / s2;
        rw[(size_t)t * NE + i2] = aff[i2] / s2;
    }
}

__global__ void add_kernel(float* __restrict__ a, const float* __restrict__ b, int n) {
    const int i = blockIdx.x * blockDim.x + threadIdx.x;
    if (i < n) a[i] += b[i];
}
__global__ void copy_kernel(float* __restrict__ dst, const float* __restrict__ src, int n) {
    const int i = blockIdx.x * blockDim.x + threadIdx.x;
    if (i < n) dst[i] = src[i];
}
__global__ void swiglu_kernel(float* __restrict__ gbuf, const float* __restrict__ ubuf, int n) {
    const int i = blockIdx.x * blockDim.x + threadIdx.x;
    if (i < n) {
        const float x = gbuf[i];
        gbuf[i] = (x / (1.0f + __expf(-x))) * ubuf[i];
    }
}
__global__ void axpy_expert_kernel(float* __restrict__ out, const float* __restrict__ ye,
                                   const float* __restrict__ rw, int e) {
    const int t = blockIdx.x;
    const float wv = rw[(size_t)t * NE + e];
    for (int h = threadIdx.x; h < Hd; h += blockDim.x)
        out[(size_t)t * Hd + h] += wv * ye[(size_t)t * Hd + h];
}

// ---------------- host launcher ----------------
extern "C" void kernel_launch(void* const* d_in, const int* in_sizes, int n_in,
                              void* d_out, int out_size, void* d_ws, size_t ws_size,
                              hipStream_t stream) {
    const float* x    = (const float*)d_in[0];
    const float* ln1  = (const float*)d_in[1];
    const float* ln2  = (const float*)d_in[2];
    const float* wq   = (const float*)d_in[3];
    const float* bq   = (const float*)d_in[4];
    const float* wk   = (const float*)d_in[5];
    const float* bk   = (const float*)d_in[6];
    const float* wv   = (const float*)d_in[7];
    const float* bv   = (const float*)d_in[8];
    const float* wo   = (const float*)d_in[9];
    const float* bo   = (const float*)d_in[10];
    const float* rwt  = (const float*)d_in[11];
    const float* gw   = (const float*)d_in[12];
    const float* uw   = (const float*)d_in[13];
    const float* dw   = (const float*)d_in[14];
    const int*   pos  = (const int*)d_in[15];
    float* out = (float*)d_out;

    const int T = in_sizes[0] / Hd;   // 1024

    // workspace layout (floats)
    float* hn  = (float*)d_ws;                 // T*Hd   (norm1 out; reused for norm2 out)
    float* q   = hn  + (size_t)T * Hd;         // T*NQD
    float* k   = q   + (size_t)T * NQD;        // T*KVD
    float* v   = k   + (size_t)T * KVD;        // T*KVD
    float* ctx = v   + (size_t)T * KVD;        // T*NQD
    float* h2  = ctx + (size_t)T * NQD;        // T*Hd
    float* rw  = h2  + (size_t)T * Hd;         // T*NE
    float* g   = rw  + (size_t)T * NE;         // T*Iin
    float* u   = g   + (size_t)T * Iin;        // T*Iin
    float* ye  = u   + (size_t)T * Iin;        // T*Hd

    const int nTH = T * Hd;
    const dim3 b256g(256), b32g(32);

    // 1) norm1
    rmsnorm_kernel<<<T, 256, 0, stream>>>(x, ln1, hn);
    // 2) QKV projections
    gemm_wmma_kernel<<<dim3(NQD / BN, T / BM), b256g, 0, stream>>>(hn, wq, bq, q, T, NQD, Hd);
    gemm_wmma_kernel<<<dim3(KVD / BN, T / BM), b256g, 0, stream>>>(hn, wk, bk, k, T, KVD, Hd);
    gemm_wmma_kernel<<<dim3(KVD / BN, T / BM), b256g, 0, stream>>>(hn, wv, bv, v, T, KVD, Hd);
    // 3) RoPE (in place on q, k)
    rope_kernel<<<T, 256, 0, stream>>>(q, k, pos);
    // 4) attention
    attn_kernel<<<dim3(T / 16, NQ), b32g, 0, stream>>>(q, k, v, ctx, T);
    // 5) output projection + residual
    gemm_wmma_kernel<<<dim3(Hd / BN, T / BM), b256g, 0, stream>>>(ctx, wo, bo, h2, T, Hd, NQD);
    add_kernel<<<(nTH + 255) / 256, b256g, 0, stream>>>(h2, x, nTH);
    // 6) norm2 (into hn)
    rmsnorm_kernel<<<T, 256, 0, stream>>>(h2, ln2, hn);
    // 7) router weights
    router_kernel<<<T, 256, 0, stream>>>(hn, rwt, rw);
    // 8) out = residual(h2); then accumulate weighted expert outputs
    copy_kernel<<<(nTH + 255) / 256, b256g, 0, stream>>>(out, h2, nTH);
    const int nTI = T * Iin;
    for (int e = 0; e < NE; ++e) {
        const float* gwe = gw + (size_t)e * Hd * Iin;
        const float* uwe = uw + (size_t)e * Hd * Iin;
        const float* dwe = dw + (size_t)e * Iin * Hd;
        gemm_wmma_kernel<<<dim3(Iin / BN, T / BM), b256g, 0, stream>>>(hn, gwe, nullptr, g, T, Iin, Hd);
        gemm_wmma_kernel<<<dim3(Iin / BN, T / BM), b256g, 0, stream>>>(hn, uwe, nullptr, u, T, Iin, Hd);
        swiglu_kernel<<<(nTI + 255) / 256, b256g, 0, stream>>>(g, u, nTI);
        gemm_wmma_kernel<<<dim3(Hd / BN, T / BM), b256g, 0, stream>>>(g, dwe, nullptr, ye, T, Hd, Iin);
        axpy_expert_kernel<<<T, 256, 0, stream>>>(out, ye, rw, e);
    }
}
